// BimanualPsiNetwork_23330262352015
// MI455X (gfx1250) — matrix operand order repaired and checked by reading, not compiled, not run on hardware
//
#include <hip/hip_runtime.h>
#include <hip/hip_bf16.h>
#include <math.h>

typedef __attribute__((ext_vector_type(16))) __bf16 v16bf;
typedef __attribute__((ext_vector_type(8)))  float  v8f;

#define NHEADS 8
#define DHEAD 32

__device__ __forceinline__ __bf16 f2bf(float f) {
  union { float f; unsigned u; } c; c.f = f;
  unsigned r = c.u + 0x7FFFu + ((c.u >> 16) & 1u);   // round-to-nearest-even
  union { unsigned short s; __bf16 b; } o;
  o.s = (unsigned short)(r >> 16);
  return o.b;
}

__device__ __forceinline__ float gelu_tanh(float x) {
  float x3 = x * x * x;
  return 0.5f * x * (1.0f + tanhf(0.7978845608028654f * (x + 0.044715f * x3)));
}

// ---------------------------------------------------------------------------
// Generic tiled GEMM: C[M,N] = act(A[M,K] @ B[K,N] + bias), optional +=.
// A: f32 M x K (converted to bf16 while staging through LDS).
// Bt: bf16, PRE-TRANSPOSED to N x K so a [col][k] tile is raw contiguous
//     bytes -> staged with CDNA5 async global->LDS DMA (ASYNCcnt).
// M%128==0, N%128==0, K%32==0. bf16 WMMA 16x16x32, f32 accumulate.
// Block: 256 threads = 8 waves (4 along M x 2 along N); wave tile 32x64.
// ---------------------------------------------------------------------------
__global__ __launch_bounds__(256)
void wmma_gemm(const float* __restrict__ A, const __bf16* __restrict__ Bt,
               float* __restrict__ C, const float* __restrict__ bias,
               int M, int N, int K, int doGelu, int doAcc)
{
  __shared__ __bf16 As[128 * 32];   // [row][k]
  __shared__ __bf16 Bs[128 * 32];   // [col][k]  (K contiguous per output col)

  const int tid  = threadIdx.x;
  const int w    = tid >> 5;
  const int lane = tid & 31;
  const int mrow = lane & 15;
  const bool lo  = lane < 16;
  const int mWave = (w >> 1) * 32;
  const int nWave = (w & 1) * 64;
  const long long mBase = (long long)blockIdx.x * 128;
  const long long nBase = (long long)blockIdx.y * 128;

  v8f acc[2][4];
  for (int i = 0; i < 2; i++)
    for (int j = 0; j < 4; j++)
      for (int r = 0; r < 8; r++) acc[i][j][r] = 0.0f;

  for (int kk = 0; kk < K; kk += 32) {
    // ---- B tile: async DMA 128 cols x 32 ks of bf16 (8KB), 16B per lane ----
    #pragma unroll
    for (int t = 0; t < 2; t++) {
      int item = tid + t * 256;            // 0..511
      int c    = item & 3;                 // 16-byte chunk within a 64B col-row
      int col  = item >> 2;                // 0..127
      const __bf16* gp = Bt + ((size_t)(nBase + col)) * K + kk + c * 8;
      unsigned lva = (unsigned)(size_t)&Bs[col * 32 + c * 8];
      unsigned long long gva = (unsigned long long)(size_t)gp;
      asm volatile("global_load_async_to_lds_b128 %0, %1, off"
                   :: "v"(lva), "v"(gva) : "memory");
    }
    // ---- A tile 128x32: f32 load, convert to bf16, ds_store ----
    #pragma unroll
    for (int t = 0; t < 4; t++) {
      int vec = tid + t * 256;
      int el  = vec * 4;
      int row = el >> 5, col = el & 31;
      const float4 f4 = *(const float4*)(A + (mBase + row) * (long long)K + kk + col);
      __bf16* d = &As[row * 32 + col];
      d[0] = f2bf(f4.x); d[1] = f2bf(f4.y); d[2] = f2bf(f4.z); d[3] = f2bf(f4.w);
    }
    asm volatile("s_wait_asynccnt 0" ::: "memory");
    __syncthreads();

    union Frag { v16bf v; unsigned u[8]; };
    Frag a[2];
    // A fragment (16x32 bf16): VGPR j -> K pair per ISA 7.12.2 16-bit A layout
    #pragma unroll
    for (int wm = 0; wm < 2; wm++) {
      int row = mWave + wm * 16 + mrow;
      const unsigned* p = (const unsigned*)&As[row * 32];
      #pragma unroll
      for (int j = 0; j < 8; j++) {
        int idx = j + ((j >= 4) ? 4 : 0) + (lo ? 0 : 4);  // pair index in row
        a[wm].u[j] = p[idx];
      }
    }
    #pragma unroll
    for (int wn = 0; wn < 4; wn++) {
      int col = nWave + wn * 16 + mrow;
      const unsigned* p = (const unsigned*)&Bs[col * 32];
      Frag b;
      #pragma unroll
      for (int j = 0; j < 8; j++) {
        int idx = j + ((j >= 4) ? 4 : 0) + (lo ? 0 : 4);  // B: same K pairs as A
        b.u[j] = p[idx];
      }
      #pragma unroll
      for (int wm = 0; wm < 2; wm++) {
        acc[wm][wn] = __builtin_amdgcn_wmma_f32_16x16x32_bf16(
            false, a[wm].v, false, b.v, (short)0, acc[wm][wn], false, false);
      }
    }
    __syncthreads();
  }

  // ---- epilogue: bias / gelu / (optional accumulate) ----
  for (int wm = 0; wm < 2; wm++) {
    for (int wn = 0; wn < 4; wn++) {
      int colN = (int)nBase + nWave + wn * 16 + mrow;
      float bv = bias ? bias[colN] : 0.0f;
      #pragma unroll
      for (int r = 0; r < 8; r++) {
        long long row = mBase + mWave + wm * 16 + (lo ? r : r + 8);
        float val = acc[wm][wn][r] + bv;
        if (doGelu) val = gelu_tanh(val);
        float* cp = C + row * (long long)N + colN;
        if (doAcc) *cp += val; else *cp = val;
      }
    }
  }
}

// B (K x N f32) -> Bt (N x K bf16), pre-transposed for the GEMM's async path
__global__ void convT_bf16(const float* __restrict__ in, __bf16* __restrict__ out,
                           int K, int N)
{
  int id = blockIdx.x * 256 + threadIdx.x;
  if (id >= N * K) return;
  int n = id / K, kq = id % K;
  out[id] = f2bf(in[(long long)kq * N + n]);
}

// ---------------------------------------------------------------------------
// CSR build (per edge set; dst-major): count -> exclusive scan -> bucket fill
// ---------------------------------------------------------------------------
__global__ void fill_f32(float* __restrict__ p, float v, long long n) {
  long long i = (long long)blockIdx.x * blockDim.x + threadIdx.x;
  if (i < n) p[i] = v;
}

__global__ void count_dst(const int* __restrict__ ei, int* __restrict__ cnt, int E) {
  int i = blockIdx.x * 256 + threadIdx.x;
  if (i < E) atomicAdd(&cnt[ei[E + i]], 1);
}

// single-block exclusive scan over N counts (N up to ~64K fine)
__global__ __launch_bounds__(256)
void scan_excl(const int* __restrict__ cnt, int* __restrict__ rowptr, int N)
{
  __shared__ int sm[256];
  __shared__ int carry;
  int tid = threadIdx.x;
  if (tid == 0) carry = 0;
  __syncthreads();
  for (int base = 0; base < N; base += 256) {
    int i = base + tid;
    int val = (i < N) ? cnt[i] : 0;
    sm[tid] = val; __syncthreads();
    for (int off = 1; off < 256; off <<= 1) {
      int t = (tid >= off) ? sm[tid - off] : 0;
      __syncthreads();
      sm[tid] += t;
      __syncthreads();
    }
    if (i < N) rowptr[i] = carry + sm[tid] - val;
    __syncthreads();
    if (tid == 255) carry += sm[255];
    __syncthreads();
  }
  if (tid == 0) rowptr[N] = carry;
}

__global__ void fill_csr(const int* __restrict__ ei, int* __restrict__ cursor,
                         int* __restrict__ eidx, int E)
{
  int i = blockIdx.x * 256 + threadIdx.x;
  if (i >= E) return;
  int pos = atomicAdd(&cursor[ei[E + i]], 1);
  eidx[pos] = i;
}

// ---------------------------------------------------------------------------
// Fused per-(dst,head) attention: score + softmax + message (no atomics).
// msg[n,h,:] = sum_e softmax(q[n,h]·(k[src]+e)) * (v[src]+e)
// ---------------------------------------------------------------------------
__global__ __launch_bounds__(256)
void attn_node(const float* __restrict__ q, const float* __restrict__ k,
               const float* __restrict__ v, const float* __restrict__ eF,
               const int* __restrict__ rowptr, const int* __restrict__ eidx,
               const int* __restrict__ ei, float* __restrict__ sbuf,
               float* __restrict__ msg, int E, int N)
{
  int idx = blockIdx.x * 256 + threadIdx.x;
  if (idx >= N * NHEADS) return;
  int n = idx >> 3, h = idx & 7;
  int beg = rowptr[n], end = rowptr[n + 1];

  float4 q4[8];
  const float4* qp = (const float4*)(q + (long long)n * 256 + h * DHEAD);
  #pragma unroll
  for (int t = 0; t < 8; t++) q4[t] = qp[t];

  // pass 1: scores + running max (stored at CSR position)
  float mx = -3.4e38f;
  for (int p = beg; p < end; ++p) {
    int e = eidx[p];
    int src = ei[e];
    const float4* kp = (const float4*)(k  + (long long)src * 256 + h * DHEAD);
    const float4* ep = (const float4*)(eF + (long long)e   * 256 + h * DHEAD);
    float acc = 0.0f;
    #pragma unroll
    for (int t = 0; t < 8; t++) {
      float4 kv = kp[t], ev = ep[t];
      acc += q4[t].x * (kv.x + ev.x) + q4[t].y * (kv.y + ev.y)
           + q4[t].z * (kv.z + ev.z) + q4[t].w * (kv.w + ev.w);
    }
    acc *= 0.17677669529663687f;   // 32^-0.5
    sbuf[(long long)p * NHEADS + h] = acc;
    mx = fmaxf(mx, acc);
  }

  // pass 2: exp, denominator, weighted aggregation in registers
  float4 a4[8];
  #pragma unroll
  for (int t = 0; t < 8; t++) a4[t] = make_float4(0.f, 0.f, 0.f, 0.f);
  float den = 0.0f;
  for (int p = beg; p < end; ++p) {
    int e = eidx[p];
    int src = ei[e];
    float ex = expf(sbuf[(long long)p * NHEADS + h] - mx);
    den += ex;
    const float4* vp = (const float4*)(v  + (long long)src * 256 + h * DHEAD);
    const float4* ep = (const float4*)(eF + (long long)e   * 256 + h * DHEAD);
    #pragma unroll
    for (int t = 0; t < 8; t++) {
      float4 vv = vp[t], ev = ep[t];
      a4[t].x += ex * (vv.x + ev.x); a4[t].y += ex * (vv.y + ev.y);
      a4[t].z += ex * (vv.z + ev.z); a4[t].w += ex * (vv.w + ev.w);
    }
  }
  float inv = 1.0f / (den + 1e-9f);
  float4* mp = (float4*)(msg + (long long)n * 256 + h * DHEAD);
  #pragma unroll
  for (int t = 0; t < 8; t++) {
    a4[t].x *= inv; a4[t].y *= inv; a4[t].z *= inv; a4[t].w *= inv;
    mp[t] = a4[t];
  }
}

// x = LayerNorm(x + a), one 256-thread block per row (H=256)
__global__ __launch_bounds__(256)
void ln_residual(float* __restrict__ x, const float* __restrict__ a)
{
  __shared__ float sm[256];
  long long row = blockIdx.x;
  int tid = threadIdx.x;
  float v = x[row * 256 + tid] + a[row * 256 + tid];
  sm[tid] = v; __syncthreads();
  for (int s2 = 128; s2 > 0; s2 >>= 1) { if (tid < s2) sm[tid] += sm[tid + s2]; __syncthreads(); }
  float mean = sm[0] * (1.0f / 256.0f);
  __syncthreads();
  float d = v - mean;
  sm[tid] = d * d; __syncthreads();
  for (int s2 = 128; s2 > 0; s2 >>= 1) { if (tid < s2) sm[tid] += sm[tid + s2]; __syncthreads(); }
  float var = sm[0] * (1.0f / 256.0f);
  x[row * 256 + tid] = d * rsqrtf(var + 1e-5f);
}

// out[i,j] = h1[i,:] @ w2[:,j] + b2[j]   (N=7, too small for WMMA)
__global__ void head_out(const float* __restrict__ h1, const float* __restrict__ w2,
                         const float* __restrict__ b2, float* __restrict__ out, int Mrows)
{
  int idx = blockIdx.x * 256 + threadIdx.x;
  if (idx >= Mrows * 7) return;
  int i = idx / 7, j = idx % 7;
  const float* hp = h1 + (long long)i * 256;
  float acc = b2[j];
  for (int c = 0; c < 256; c++) acc += hp[c] * w2[c * 7 + j];
  out[idx] = acc;
}

// ---------------------------------------------------------------------------
// Host orchestration
// ---------------------------------------------------------------------------
namespace {

struct Ws {
  float *xl, *xr, *q, *k, *v, *msg, *aggL, *aggR, *tmp, *sbuf, *eF;
  __bf16* bfB;
  int* cnt;                       // scratch counts / cursor (N ints)
  int* rowptr[4];                 // per edge set, N+1 ints
  int* eidx[4];                   // per edge set, E ints
  int N, EA, ES;
};

inline void run_gemm(hipStream_t st, const Ws& W, const float* A, const float* B,
                     float* C, const float* bias, int M, int N, int K,
                     int gelu, int acc) {
  convT_bf16<<<(unsigned)((N * K + 255) / 256), 256, 0, st>>>(B, W.bfB, K, N);
  dim3 grid((unsigned)(M / 128), (unsigned)(N / 128));
  wmma_gemm<<<grid, 256, 0, st>>>(A, W.bfB, C, bias, M, N, K, gelu, acc);
}

inline void build_csr(hipStream_t st, const Ws& W, const int* ei, int E,
                      int* rowptr, int* eidx) {
  fill_f32<<<(unsigned)((W.N + 255) / 256), 256, 0, st>>>((float*)W.cnt, 0.0f, W.N);
  count_dst<<<(unsigned)((E + 255) / 256), 256, 0, st>>>(ei, W.cnt, E);
  scan_excl<<<1, 256, 0, st>>>(W.cnt, rowptr, W.N);
  hipMemcpyAsync(W.cnt, rowptr, (size_t)W.N * sizeof(int),
                 hipMemcpyDeviceToDevice, st);
  fill_csr<<<(unsigned)((E + 255) / 256), 256, 0, st>>>(ei, W.cnt, eidx, E);
}

inline void run_attn(hipStream_t st, const Ws& W, const float* xq, const float* xk,
                     const int* ei, const float* ea, int E, int set,
                     const float* wq, const float* wk, const float* wv,
                     const float* we, const float* wo, float* agg) {
  const int N = W.N;
  run_gemm(st, W, xq, wq, W.q,  nullptr, N, 256, 256, 0, 0);
  run_gemm(st, W, xk, wk, W.k,  nullptr, N, 256, 256, 0, 0);
  run_gemm(st, W, xk, wv, W.v,  nullptr, N, 256, 256, 0, 0);
  run_gemm(st, W, ea, we, W.eF, nullptr, E, 256,  64, 0, 0);

  attn_node<<<(unsigned)((N * NHEADS + 255) / 256), 256, 0, st>>>(
      W.q, W.k, W.v, W.eF, W.rowptr[set], W.eidx[set], ei, W.sbuf, W.msg, E, N);

  run_gemm(st, W, W.msg, wo, agg, nullptr, N, 256, 256, 0, /*acc=*/1);
}

} // namespace

extern "C" void kernel_launch(void* const* d_in, const int* in_sizes, int n_in,
                              void* d_out, int out_size, void* d_ws, size_t ws_size,
                              hipStream_t stream) {
  const float* afl   = (const float*)d_in[0];
  const float* afr   = (const float*)d_in[1];
  const float* ea_ll = (const float*)d_in[2];
  const float* ea_rr = (const float*)d_in[3];
  const float* ea_lr = (const float*)d_in[4];
  const float* ea_rl = (const float*)d_in[5];
  const int*   ei_ll = (const int*)d_in[6];
  const int*   ei_rr = (const int*)d_in[7];
  const int*   ei_lr = (const int*)d_in[8];
  const int*   ei_rl = (const int*)d_in[9];
  const float* Wq  = (const float*)d_in[10];
  const float* Wk  = (const float*)d_in[11];
  const float* Wv  = (const float*)d_in[12];
  const float* We  = (const float*)d_in[13];
  const float* Wo  = (const float*)d_in[14];
  const float* fw1 = (const float*)d_in[15];
  const float* fb1 = (const float*)d_in[16];
  const float* fw2 = (const float*)d_in[17];
  const float* fb2 = (const float*)d_in[18];
  const float* hw1 = (const float*)d_in[19];
  const float* hb1 = (const float*)d_in[20];
  const float* hw2 = (const float*)d_in[21];
  const float* hb2 = (const float*)d_in[22];

  const int N  = in_sizes[0] / 256;
  const int EA = in_sizes[2] / 64;
  const int ES = in_sizes[4] / 64;
  const int L  = in_sizes[10] / (4 * 256 * 256);
  const int AS = 2048;                 // action_start (fixed by harness setup)
  const int Mh = N - AS;

  // ---- workspace carve (256B aligned blocks) ----
  char* base = (char*)d_ws;
  auto alloc = [&](size_t bytes) -> void* {
    void* r = (void*)base;
    base += (bytes + 255) & ~(size_t)255;
    return r;
  };
  Ws W; W.N = N; W.EA = EA; W.ES = ES;
  W.xl   = (float*)alloc((size_t)N * 256 * 4);
  W.xr   = (float*)alloc((size_t)N * 256 * 4);
  W.q    = (float*)alloc((size_t)N * 256 * 4);
  W.k    = (float*)alloc((size_t)N * 256 * 4);
  W.v    = (float*)alloc((size_t)N * 256 * 4);
  W.msg  = (float*)alloc((size_t)N * 256 * 4);
  W.aggL = (float*)alloc((size_t)N * 256 * 4);
  W.aggR = (float*)alloc((size_t)N * 256 * 4);
  W.tmp  = (float*)alloc((size_t)N * 512 * 4);
  W.sbuf = (float*)alloc((size_t)EA * NHEADS * 4);
  W.bfB  = (__bf16*)alloc((size_t)512 * 256 * 2);
  W.cnt  = (int*)alloc((size_t)N * 4);
  const int Es[4] = {EA, EA, ES, ES};
  for (int sidx = 0; sidx < 4; sidx++) {
    W.rowptr[sidx] = (int*)alloc((size_t)(N + 1) * 4);
    W.eidx[sidx]   = (int*)alloc((size_t)Es[sidx] * 4);
  }
  W.eF = (float*)alloc((size_t)EA * 256 * 4);

  // ---- CSR for each edge set (dst-major); reused across layers ----
  build_csr(stream, W, ei_ll, EA, W.rowptr[0], W.eidx[0]);
  build_csr(stream, W, ei_rr, EA, W.rowptr[1], W.eidx[1]);
  build_csr(stream, W, ei_lr, ES, W.rowptr[2], W.eidx[2]);
  build_csr(stream, W, ei_rl, ES, W.rowptr[3], W.eidx[3]);

  hipMemcpyAsync(W.xl, afl, (size_t)N * 256 * sizeof(float), hipMemcpyDeviceToDevice, stream);
  hipMemcpyAsync(W.xr, afr, (size_t)N * 256 * sizeof(float), hipMemcpyDeviceToDevice, stream);

  const long long WQK = 256 * 256, WEK = 64 * 256;
  const long long F1 = 256 * 512, F2 = 512 * 256;
  const long long nmsg = (long long)N * 256;

  for (int l = 0; l < L; l++) {
    fill_f32<<<(unsigned)((nmsg + 255) / 256), 256, 0, stream>>>(W.aggL, 0.0f, nmsg);
    fill_f32<<<(unsigned)((nmsg + 255) / 256), 256, 0, stream>>>(W.aggR, 0.0f, nmsg);

    // agg_l = attn(xl, xl, ll/set0, W[l,0]) + attn(xl, xr, rl/set3, W[l,3])
    run_attn(stream, W, W.xl, W.xl, ei_ll, ea_ll, EA, 0,
             Wq + (l * 4 + 0) * WQK, Wk + (l * 4 + 0) * WQK, Wv + (l * 4 + 0) * WQK,
             We + (l * 4 + 0) * WEK, Wo + (l * 4 + 0) * WQK, W.aggL);
    run_attn(stream, W, W.xl, W.xr, ei_rl, ea_rl, ES, 3,
             Wq + (l * 4 + 3) * WQK, Wk + (l * 4 + 3) * WQK, Wv + (l * 4 + 3) * WQK,
             We + (l * 4 + 3) * WEK, Wo + (l * 4 + 3) * WQK, W.aggL);
    // agg_r = attn(xr, xr, rr/set1, W[l,1]) + attn(xr, xl, lr/set2, W[l,2])
    run_attn(stream, W, W.xr, W.xr, ei_rr, ea_rr, EA, 1,
             Wq + (l * 4 + 1) * WQK, Wk + (l * 4 + 1) * WQK, Wv + (l * 4 + 1) * WQK,
             We + (l * 4 + 1) * WEK, Wo + (l * 4 + 1) * WQK, W.aggR);
    run_attn(stream, W, W.xr, W.xl, ei_lr, ea_lr, ES, 2,
             Wq + (l * 4 + 2) * WQK, Wk + (l * 4 + 2) * WQK, Wv + (l * 4 + 2) * WQK,
             We + (l * 4 + 2) * WEK, Wo + (l * 4 + 2) * WQK, W.aggR);

    ln_residual<<<(unsigned)N, 256, 0, stream>>>(W.xl, W.aggL);
    ln_residual<<<(unsigned)N, 256, 0, stream>>>(W.xr, W.aggR);

    // FFN: x = LN(x + W2(gelu(W1 x + b1)) + b2)
    run_gemm(stream, W, W.xl, fw1 + (l * 2 + 0) * F1, W.tmp, fb1 + (l * 2 + 0) * 512,
             N, 512, 256, /*gelu=*/1, 0);
    run_gemm(stream, W, W.tmp, fw2 + (l * 2 + 0) * F2, W.msg, fb2 + (l * 2 + 0) * 256,
             N, 256, 512, 0, 0);
    ln_residual<<<(unsigned)N, 256, 0, stream>>>(W.xl, W.msg);

    run_gemm(stream, W, W.xr, fw1 + (l * 2 + 1) * F1, W.tmp, fb1 + (l * 2 + 1) * 512,
             N, 512, 256, /*gelu=*/1, 0);
    run_gemm(stream, W, W.tmp, fw2 + (l * 2 + 1) * F2, W.msg, fb2 + (l * 2 + 1) * 256,
             N, 256, 512, 0, 0);
    ln_residual<<<(unsigned)N, 256, 0, stream>>>(W.xr, W.msg);
  }

  // head on the action slice
  float* out = (float*)d_out;
  run_gemm(stream, W, W.xl + (long long)AS * 256, hw1, W.tmp, hb1, Mh, 256, 256, 1, 0);
  head_out<<<(unsigned)((Mh * 7 + 255) / 256), 256, 0, stream>>>(W.tmp, hw2, hb2, out, Mh);
  run_gemm(stream, W, W.xr + (long long)AS * 256, hw1, W.tmp, hb1, Mh, 256, 256, 1, 0);
  head_out<<<(unsigned)((Mh * 7 + 255) / 256), 256, 0, stream>>>(W.tmp, hw2, hb2,
                                                                 out + (long long)Mh * 7, Mh);
}
